// RNN_autoreg_22204980920431
// MI455X (gfx1250) — matrix operand order repaired and checked by reading, not compiled, not run on hardware
//
#include <hip/hip_runtime.h>
#include <stdint.h>

#define BATCH 16384
#define TSEQ  60
#define NXF   4
#define NXSF  3
#define NYF   4
#define NYSF  3
#define HDIM  128
#define G4H   512     // 4*H gate width
#define BT    16      // batch rows per block
#define NTHR  256     // 8 wave32

// LSTM1: K = [mem(128) | h(128) | x(4) | 1.0 | 0-pad] = 288  (bias folded via ones col)
#define K1    288
#define NKC1  9
#define WSTR1 296     // halves/row: 592B (16B-mult, 148 dwords -> 4-way banks)
#define A1STR 296
// LSTM2: K = [r1(128) | h(128)] = 256, bias splat into accumulator
#define K2    256
#define NKC2  8
#define WSTR2 264     // 528B
#define A2STR 264
#define WLSTR 136     // Wlat row: 272B
#define LSTR  132

// ---- LDS layouts (bytes) ----
// LSTM1: W[512][296]bf16 | A[16][296]bf16
#define OFF1_A    303104
#define SMEM1     312576
// LSTM2: W[512][264] | A[16][264] | bias[512]f32 | Wlat[128][136]bf16 | lat[16][132]f32
//        | Wout[4][128]f32 | blat[128]f32 | bout[4]f32
#define OFF2_A    270336
#define OFF2_BIAS 278784
#define OFF2_WLAT 280832
#define OFF2_LAT  315648
#define OFF2_WOUT 324096
#define OFF2_BLAT 326144
#define OFF2_BOUT 326656
#define SMEM2     326672

typedef __attribute__((ext_vector_type(16))) __bf16 v16bf;
typedef __attribute__((ext_vector_type(8)))  __bf16 v8bf;
typedef __attribute__((ext_vector_type(8)))  float  v8f;
typedef __attribute__((ext_vector_type(4)))  float  v4f;

union Frag  { v16bf v; v8bf h[2]; };
union Pack8 { v8bf  v; __bf16 e[8]; };

__device__ __forceinline__ __bf16 f2bf(float f) {
  union { float f; uint32_t u; } x; x.f = f;
  uint32_t r = x.u + 0x7FFFu + ((x.u >> 16) & 1u);   // round-to-nearest-even
  uint16_t hb = (uint16_t)(r >> 16);
  __bf16 b; __builtin_memcpy(&b, &hb, 2); return b;
}
__device__ __forceinline__ float bf2f(__bf16 b) {
  uint16_t hb; __builtin_memcpy(&hb, &b, 2);
  union { uint32_t u; float f; } x; x.u = ((uint32_t)hb) << 16; return x.f;
}
__device__ __forceinline__ float sigmoidf_(float x) { return 1.f / (1.f + expf(-x)); }

// A fragment (16x32 bf16, MxK), ISA 7.12.2: lane<16 -> m=lane, k=kb+{0..7,16..23};
// lane>=16 -> m=lane-16, k=kb+{8..15,24..31}.
__device__ __forceinline__ v16bf load_a_frag(const __bf16* A, int astr, int lane, int kb) {
  int m  = lane & 15;
  int ko = kb + ((lane >> 4) << 3);
  Frag f;
  f.h[0] = *(const v8bf*)(A + m * astr + ko);
  f.h[1] = *(const v8bf*)(A + m * astr + ko + 16);
  return f.v;
}
// B fragment (32x16 bf16, KxN) from row-major W[n][k] (B = W^T):
// lane<16 -> n=nb+lane, k=kb+{0..7,16..23}; lane>=16 -> n=nb+lane-16, k=kb+{8..15,24..31}.
__device__ __forceinline__ v16bf load_b_frag(const __bf16* W, int wstr, int lane, int nb, int kb) {
  int n  = nb + (lane & 15);
  int ko = kb + ((lane >> 4) << 3);
  const __bf16* p = W + n * wstr + ko;
  Frag f; f.h[0] = *(const v8bf*)p; f.h[1] = *(const v8bf*)(p + 16);
  return f.v;
}

// =============================== LSTM layer 1 ===============================
__global__ __launch_bounds__(NTHR)
void k_lstm1(const float* __restrict__ xmain, const float* __restrict__ sfc,
             const float* __restrict__ mem0,
             const float* __restrict__ Ws1, const float* __restrict__ bs1,
             const float* __restrict__ Ws2, const float* __restrict__ bs2,
             const float* __restrict__ Wih1, const float* __restrict__ Whh1,
             const float* __restrict__ bih1, const float* __restrict__ bhh1,
             __bf16* __restrict__ r1out)
{
  extern __shared__ char smem[];
  __bf16* Wl   = (__bf16*)(smem);
  __bf16* Abuf = (__bf16*)(smem + OFF1_A);

  const int tid  = threadIdx.x;
  const int lane = tid & 31;
  const int wv   = tid >> 5;
  const int b0   = blockIdx.x * BT;
  const int nh   = (wv << 4) + (lane & 15);
  const int mhi  = (lane >> 4) << 3;

  // --- pack combined weights [Wih1(mem) | Whh1 | Wih1(x) | bias | 0] as bf16 ---
  for (int row = tid; row < G4H; row += NTHR) {
    __bf16* wr = Wl + row * WSTR1;
    const float* ih = Wih1 + row * (NXF + HDIM);
    const float* hh = Whh1 + row * HDIM;
    for (int k = 0; k < HDIM; ++k) wr[k]        = f2bf(ih[NXF + k]);
    for (int k = 0; k < HDIM; ++k) wr[HDIM + k] = f2bf(hh[k]);
    for (int j = 0; j < NXF;  ++j) wr[256 + j]  = f2bf(ih[j]);
    wr[260] = f2bf(bih1[row] + bhh1[row]);
    for (int k = 261; k < K1; ++k) wr[k] = f2bf(0.f);
  }
  // --- A constant tail: ones column (bias) + zero pad ---
  if (tid < BT) {
    for (int k = 256; k < K1; ++k)
      Abuf[tid * A1STR + k] = (k == 260) ? f2bf(1.0f) : f2bf(0.0f);
  }
  // --- h0 = tanh(sfc @ Ws1^T + bs1) -> A[:,128:256) ---
  {
    int e = tid * 8;
    int m = e >> 7, n0 = e & 127;
    const float* sp = sfc + (size_t)(b0 + m) * NXSF;
    Pack8 pk;
    #pragma unroll
    for (int q = 0; q < 8; ++q) {
      const float* w = Ws1 + (n0 + q) * NXSF;
      pk.e[q] = f2bf(tanhf(bs1[n0 + q] + sp[0]*w[0] + sp[1]*w[1] + sp[2]*w[2]));
    }
    *(v8bf*)(Abuf + m * A1STR + HDIM + n0) = pk.v;
  }
  // --- c0 = tanh(sfc @ Ws2^T + bs2), fp32 fragment in registers ---
  float c[8];
  {
    const float* w = Ws2 + nh * NXSF;
    #pragma unroll
    for (int r = 0; r < 8; ++r) {
      const float* sp = sfc + (size_t)(b0 + r + mhi) * NXSF;
      c[r] = tanhf(bs2[nh] + sp[0]*w[0] + sp[1]*w[1] + sp[2]*w[2]);
    }
  }
  __syncthreads();

  for (int t = 0; t < TSEQ; ++t) {
    // stage mem0[:,t,:] -> A[:,0:128) (fp32 -> bf16), x_t -> A[:,256:260)
    {
      int m = tid >> 4, k0 = (tid & 15) * 8;
      const float* gp = mem0 + ((size_t)(b0 + m) * TSEQ + t) * HDIM + k0;
      v4f a = *(const v4f*)gp;
      v4f b = *(const v4f*)(gp + 4);
      Pack8 pk;
      pk.e[0]=f2bf(a.x); pk.e[1]=f2bf(a.y); pk.e[2]=f2bf(a.z); pk.e[3]=f2bf(a.w);
      pk.e[4]=f2bf(b.x); pk.e[5]=f2bf(b.y); pk.e[6]=f2bf(b.z); pk.e[7]=f2bf(b.w);
      *(v8bf*)(Abuf + m * A1STR + k0) = pk.v;
      if (t + 1 < TSEQ) __builtin_prefetch(gp + HDIM, 0, 1);   // global_prefetch_b8
    }
    if (tid < BT * NXF) {
      int m = tid >> 2, j = tid & 3;
      Abuf[m * A1STR + 256 + j] =
          f2bf(xmain[((size_t)(b0 + m) * TSEQ + (TSEQ - 1 - t)) * NXF + j]);
    }
    __syncthreads();

    // z = [mem | h | x | 1] @ W^T  -- software-pipelined fragment loads
    v8f acc[4];
    v8f zro = {};
    v16bf a = load_a_frag(Abuf, A1STR, lane, 0);
    v16bf b[4];
    #pragma unroll
    for (int g = 0; g < 4; ++g)
      b[g] = load_b_frag(Wl, WSTR1, lane, g * HDIM + (wv << 4), 0);
    #pragma unroll
    for (int kc = 0; kc < NKC1; ++kc) {
      v16bf an = a;
      v16bf bn[4] = { b[0], b[1], b[2], b[3] };
      if (kc + 1 < NKC1) {
        an = load_a_frag(Abuf, A1STR, lane, (kc + 1) * 32);
        #pragma unroll
        for (int g = 0; g < 4; ++g)
          bn[g] = load_b_frag(Wl, WSTR1, lane, g * HDIM + (wv << 4), (kc + 1) * 32);
      }
      #pragma unroll
      for (int g = 0; g < 4; ++g)
        acc[g] = __builtin_amdgcn_wmma_f32_16x16x32_bf16(false, a, false, b[g],
                          (short)0, (kc == 0) ? zro : acc[g], false, false);
      a = an;
      #pragma unroll
      for (int g = 0; g < 4; ++g) b[g] = bn[g];
    }
    __syncthreads();                         // all A reads done
    asm volatile("s_wait_asynccnt 0x0" ::: "memory");  // prev r1 store drained

    // pointwise LSTM cell; h -> A[:,128:256)
    #pragma unroll
    for (int r = 0; r < 8; ++r) {
      int m = r + mhi;
      float ig = sigmoidf_(acc[0][r]);
      float fg = sigmoidf_(acc[1][r]);
      float gg = tanhf(acc[2][r]);
      float og = sigmoidf_(acc[3][r]);
      c[r] = fg * c[r] + ig * gg;
      Abuf[m * A1STR + HDIM + nh] = f2bf(og * tanhf(c[r]));
    }
    __syncthreads();

    // r1[:,T-1-t,:] = h_t : async DMA straight from LDS (ASYNCcnt-tracked)
    {
      int m = tid >> 4, k0 = (tid & 15) * 8;
      uint32_t lds = (uint32_t)(OFF1_A + (m * A1STR + HDIM + k0) * 2);
      const __bf16* gp = r1out + ((size_t)(b0 + m) * TSEQ + (TSEQ - 1 - t)) * HDIM + k0;
      asm volatile("global_store_async_from_lds_b128 %0, %1, off"
                   :: "v"(gp), "v"(lds) : "memory");
    }
    __syncthreads();
  }
  // s_endpgm performs an implicit wait-idle for outstanding async stores
}

// =============================== LSTM layer 2 + heads =======================
__global__ __launch_bounds__(NTHR)
void k_lstm2(const __bf16* __restrict__ r1, const float* __restrict__ hx2,
             const float* __restrict__ cx2,
             const float* __restrict__ Wih2, const float* __restrict__ Whh2,
             const float* __restrict__ bih2, const float* __restrict__ bhh2,
             const float* __restrict__ Wlat, const float* __restrict__ blat,
             const float* __restrict__ Wout, const float* __restrict__ bout,
             const float* __restrict__ Wsfc, const float* __restrict__ bsfc,
             float* __restrict__ out, float* __restrict__ outsfc)
{
  extern __shared__ char smem[];
  __bf16* Wl    = (__bf16*)(smem);
  __bf16* Abuf  = (__bf16*)(smem + OFF2_A);
  float*  bias  = (float*)(smem + OFF2_BIAS);
  __bf16* Wlatl = (__bf16*)(smem + OFF2_WLAT);
  float*  latb  = (float*)(smem + OFF2_LAT);
  float*  woutl = (float*)(smem + OFF2_WOUT);
  float*  blatl = (float*)(smem + OFF2_BLAT);
  float*  boutl = (float*)(smem + OFF2_BOUT);

  const int tid  = threadIdx.x;
  const int lane = tid & 31;
  const int wv   = tid >> 5;
  const int b0   = blockIdx.x * BT;
  const int nh   = (wv << 4) + (lane & 15);
  const int mhi  = (lane >> 4) << 3;

  for (int idx = tid; idx < G4H * K2; idx += NTHR) {
    int n = idx >> 8, k = idx & 255;
    float w = (k < HDIM) ? Wih2[n * HDIM + k] : Whh2[n * HDIM + (k - HDIM)];
    Wl[n * WSTR2 + k] = f2bf(w);
  }
  for (int n = tid; n < G4H; n += NTHR) bias[n] = bih2[n] + bhh2[n];
  for (int idx = tid; idx < HDIM * HDIM; idx += NTHR) {
    int n = idx >> 7, k = idx & 127;
    Wlatl[n * WLSTR + k] = f2bf(Wlat[idx]);
  }
  for (int idx = tid; idx < NYF * HDIM; idx += NTHR) woutl[idx] = Wout[idx];
  for (int n = tid; n < HDIM; n += NTHR) blatl[n] = blat[n];
  if (tid < NYF) boutl[tid] = bout[tid];

  // h0 = hx2 -> A[:,128:256), c0 = cx2 fragment
  {
    int e = tid * 8;
    int m = e >> 7, n0 = e & 127;
    const float* gp = hx2 + (size_t)(b0 + m) * HDIM + n0;
    Pack8 pk;
    #pragma unroll
    for (int q = 0; q < 8; ++q) pk.e[q] = f2bf(gp[q]);
    *(v8bf*)(Abuf + m * A2STR + HDIM + n0) = pk.v;
  }
  float c[8];
  #pragma unroll
  for (int r = 0; r < 8; ++r) c[r] = cx2[(size_t)(b0 + r + mhi) * HDIM + nh];

  // issue async load of r1[:,0,:] -> A[:,0:128)
  {
    int m = tid >> 4, k0 = (tid & 15) * 8;
    uint32_t lds = (uint32_t)(OFF2_A + (m * A2STR + k0) * 2);
    const __bf16* gp = r1 + (size_t)(b0 + m) * TSEQ * HDIM + k0;
    asm volatile("global_load_async_to_lds_b128 %0, %1, off"
                 :: "v"(lds), "v"(gp) : "memory");
  }

  for (int t = 0; t < TSEQ; ++t) {
    asm volatile("s_wait_asynccnt 0x0" ::: "memory");  // r1 tile landed in LDS
    __syncthreads();

    // z = [r1_t | h] @ W^T + b  -- pipelined fragment loads
    v8f acc[4];
    #pragma unroll
    for (int g = 0; g < 4; ++g) {
      float base = bias[g * HDIM + nh];
      #pragma unroll
      for (int r = 0; r < 8; ++r) acc[g][r] = base;
    }
    v16bf a = load_a_frag(Abuf, A2STR, lane, 0);
    v16bf b[4];
    #pragma unroll
    for (int g = 0; g < 4; ++g)
      b[g] = load_b_frag(Wl, WSTR2, lane, g * HDIM + (wv << 4), 0);
    #pragma unroll
    for (int kc = 0; kc < NKC2; ++kc) {
      v16bf an = a;
      v16bf bn[4] = { b[0], b[1], b[2], b[3] };
      if (kc + 1 < NKC2) {
        an = load_a_frag(Abuf, A2STR, lane, (kc + 1) * 32);
        #pragma unroll
        for (int g = 0; g < 4; ++g)
          bn[g] = load_b_frag(Wl, WSTR2, lane, g * HDIM + (wv << 4), (kc + 1) * 32);
      }
      #pragma unroll
      for (int g = 0; g < 4; ++g)
        acc[g] = __builtin_amdgcn_wmma_f32_16x16x32_bf16(false, a, false, b[g],
                                                         (short)0, acc[g], false, false);
      a = an;
      #pragma unroll
      for (int g = 0; g < 4; ++g) b[g] = bn[g];
    }
    __syncthreads();                          // input region free

    // overlap next tile's DMA with pointwise + lat + out
    if (t + 1 < TSEQ) {
      int m = tid >> 4, k0 = (tid & 15) * 8;
      uint32_t lds = (uint32_t)(OFF2_A + (m * A2STR + k0) * 2);
      const __bf16* gp = r1 + ((size_t)(b0 + m) * TSEQ + t + 1) * HDIM + k0;
      asm volatile("global_load_async_to_lds_b128 %0, %1, off"
                   :: "v"(lds), "v"(gp) : "memory");
    }

    #pragma unroll
    for (int r = 0; r < 8; ++r) {
      int m = r + mhi;
      float ig = sigmoidf_(acc[0][r]);
      float fg = sigmoidf_(acc[1][r]);
      float gg = tanhf(acc[2][r]);
      float og = sigmoidf_(acc[3][r]);
      c[r] = fg * c[r] + ig * gg;
      Abuf[m * A2STR + HDIM + nh] = f2bf(og * tanhf(c[r]));
    }
    __syncthreads();

    // lat = h_t @ Wlat^T + blat
    {
      v8f lacc;
      float bl = blatl[nh];
      #pragma unroll
      for (int r = 0; r < 8; ++r) lacc[r] = bl;
      #pragma unroll
      for (int kc = 0; kc < HDIM / 32; ++kc) {
        v16bf al = load_a_frag(Abuf + HDIM, A2STR, lane, kc * 32);
        v16bf bl2 = load_b_frag(Wlatl, WLSTR, lane, wv << 4, kc * 32);
        lacc = __builtin_amdgcn_wmma_f32_16x16x32_bf16(false, al, false, bl2,
                                                       (short)0, lacc, false, false);
      }
      #pragma unroll
      for (int r = 0; r < 8; ++r) latb[(r + mhi) * LSTR + nh] = lacc[r];
    }
    __syncthreads();

    // out[:,t,:] = lat @ Wout^T + bout
    if (tid < BT * NYF) {
      int m = tid >> 2, y = tid & 3;
      float s = boutl[y];
      const float* lp = latb + m * LSTR;
      const float* wp = woutl + y * HDIM;
      #pragma unroll 8
      for (int n = 0; n < HDIM; ++n) s += lp[n] * wp[n];
      out[((size_t)(b0 + m) * TSEQ + t) * NYF + y] = s;
    }
    // next iteration's top wait+barrier orders latb reuse and A overwrite
  }

  // out_sfc = hN @ Wsfc^T + bsfc
  if (tid < BT * NYSF) {
    int m = tid / NYSF, y = tid % NYSF;
    float s = bsfc[y];
    const float* wp = Wsfc + y * HDIM;
    const __bf16* hp = Abuf + m * A2STR + HDIM;
    #pragma unroll 8
    for (int n = 0; n < HDIM; ++n) s += bf2f(hp[n]) * wp[n];
    outsfc[(size_t)(b0 + m) * NYSF + y] = s;
  }
}

extern "C" void kernel_launch(void* const* d_in, const int* in_sizes, int n_in,
                              void* d_out, int out_size, void* d_ws, size_t ws_size,
                              hipStream_t stream) {
  const float* inputs_main = (const float*)d_in[0];
  const float* inputs_sfc  = (const float*)d_in[1];
  const float* rnn1_mem0   = (const float*)d_in[2];
  const float* hx2  = (const float*)d_in[3];
  const float* cx2  = (const float*)d_in[4];
  const float* Ws1  = (const float*)d_in[5];
  const float* bs1  = (const float*)d_in[6];
  const float* Ws2  = (const float*)d_in[7];
  const float* bs2  = (const float*)d_in[8];
  const float* Wih1 = (const float*)d_in[9];
  const float* Whh1 = (const float*)d_in[10];
  const float* bih1 = (const float*)d_in[11];
  const float* bhh1 = (const float*)d_in[12];
  const float* Wih2 = (const float*)d_in[13];
  const float* Whh2 = (const float*)d_in[14];
  const float* bih2 = (const float*)d_in[15];
  const float* bhh2 = (const float*)d_in[16];
  const float* Wlat = (const float*)d_in[17];
  const float* blat = (const float*)d_in[18];
  const float* Wout = (const float*)d_in[19];
  const float* bout = (const float*)d_in[20];
  const float* Wsfc = (const float*)d_in[21];
  const float* bsfc = (const float*)d_in[22];

  __bf16* r1  = (__bf16*)d_ws;                     // [B][T][H] bf16 intermediate
  float*  out = (float*)d_out;                     // [B][T][NY]
  float*  osf = out + (size_t)BATCH * TSEQ * NYF;  // [B][NYS]

  dim3 grid(BATCH / BT), block(NTHR);
  k_lstm1<<<grid, block, SMEM1, stream>>>(inputs_main, inputs_sfc, rnn1_mem0,
                                          Ws1, bs1, Ws2, bs2,
                                          Wih1, Whh1, bih1, bhh1, r1);
  k_lstm2<<<grid, block, SMEM2, stream>>>(r1, hx2, cx2,
                                          Wih2, Whh2, bih2, bhh2,
                                          Wlat, blat, Wout, bout, Wsfc, bsfc,
                                          out, osf);
}
